// LabelModel_Theta_65575560675889
// MI455X (gfx1250) — compile-verified, matrix operand order checked by reading, not run on hardware
//
#include <hip/hip_runtime.h>

#define NROWS 200000
#define KDIM 256
#define KC 32                      // K per WMMA step
#define TILE_ROWS 16
#define NTILES (NROWS / TILE_ROWS) // 12500, exact
#define NCHUNKS (KDIM / KC)        // 8
#define WAVES_PER_BLOCK 8
#define BLOCK 256
#define NBLOCKS 512
#define PSTRIDE 36                 // padded floats per row in LDS stage (bank-conflict-free)

typedef _Float16 v16h __attribute__((ext_vector_type(16)));
typedef _Float16 v8h  __attribute__((ext_vector_type(8)));
typedef float    v8f  __attribute__((ext_vector_type(8)));
typedef float    v4f  __attribute__((ext_vector_type(4)));
typedef int      v4i  __attribute__((ext_vector_type(4)));

#define GAS __attribute__((address_space(1)))
#define LAS __attribute__((address_space(3)))

// Async-copy one 16x32 f32 chunk (2KB) global -> LDS, 4 x b128 per lane.
// Rows land at padded stride PSTRIDE to avoid bank conflicts on the
// per-row fragment gathers that follow.
__device__ __forceinline__ void async_chunk_load(const float* gsrc,
                                                 float* lds_dst, int lane) {
#pragma unroll
  for (int j = 0; j < 4; ++j) {
    int f   = lane + 32 * j;       // float4 index in [0,128): 8 per row
    int row = f >> 3;
    int c4  = (f & 7) * 4;
    __builtin_amdgcn_global_load_async_to_lds_b128(
        (GAS v4i*)(gsrc + row * KDIM + c4),
        (LAS v4i*)(lds_dst + row * PSTRIDE + c4), 0, 0);
  }
}

__global__ __launch_bounds__(BLOCK) void lm_tile_kernel(
    const float* __restrict__ Y, const float* __restrict__ theta,
    float* __restrict__ blockPartials) {
  __shared__ __align__(16) float stage[WAVES_PER_BLOCK][2][TILE_ROWS * PSTRIDE];
  __shared__ __align__(16) _Float16 alph[KDIM];
  __shared__ __align__(16) _Float16 alph2[KDIM];
  __shared__ float red[WAVES_PER_BLOCK];

  const int tid = threadIdx.x;
  // alpha_k = 1 - sigmoid(theta-5) = 1/(1+exp(theta-5))
  if (tid < KDIM) {
    float a = 1.0f / (1.0f + __expf(theta[tid] - 5.0f));
    alph[tid]  = (_Float16)a;
    alph2[tid] = (_Float16)(a * a);
  }
  __syncthreads();

  const int wave   = tid >> 5;
  const int lane   = tid & 31;
  const int col    = lane & 15;   // N-column this lane holds in B/D fragments
  const int half16 = lane >> 4;   // lane group 0/1

  // Constant B fragment for the mask chain: column 0 = all ones.
  v16h bOnes;
  {
    _Float16 v = (col == 0) ? (_Float16)1.0f : (_Float16)0.0f;
#pragma unroll
    for (int i = 0; i < 16; ++i) bOnes[i] = v;
  }

  const _Float16 h1 = (_Float16)1.0f;
  const _Float16 h0 = (_Float16)0.0f;

  float waveAcc = 0.0f;
  const int waveGlobal = blockIdx.x * WAVES_PER_BLOCK + wave;
  const int waveStride = gridDim.x * WAVES_PER_BLOCK;

  for (int tile = waveGlobal; tile < NTILES; tile += waveStride) {
    const float* ytile = Y + (size_t)tile * TILE_ROWS * KDIM;

    // hide next tile's first-chunk latency a bit (chunk 0 is 2KB)
    if (tile + waveStride < NTILES)
      __builtin_prefetch(Y + (size_t)(tile + waveStride) * TILE_ROWS * KDIM +
                             lane * 16, 0, 0);

    v8f accP = {};   // D col0 = S1, col1 = S2 (per row)
    v8f accM = {};   // D col0 = cnt (per row)

    // prime the double buffer
    async_chunk_load(ytile, stage[wave][0], lane);

#pragma unroll
    for (int kbi = 0; kbi < NCHUNKS; ++kbi) {
      const int kb = kbi * KC;
      if (kbi + 1 < NCHUNKS) {
        async_chunk_load(ytile + kb + KC, stage[wave][(kbi + 1) & 1], lane);
        // 8 async loads in flight; wait until the older 4 (this chunk) land
        asm volatile("s_wait_asynccnt 4" ::: "memory");
      } else {
        asm volatile("s_wait_asynccnt 0" ::: "memory");
      }

      const float* buf = stage[wave][kbi & 1];

      // ---- gather this lane's A-fragment elements straight from LDS ----
      // 16-bit A 16x32 layout: lanes 0-15: M=lane, K=0..7 (v0-3), 16..23 (v4-7)
      //                        lanes16-31: M=lane-16, K=8..15, 24..31
      const int arow = col;
      const int ko   = 8 * half16;
      const float* rowp = buf + arow * PSTRIDE;
      v4f y0 = *(const v4f*)(rowp + ko);
      v4f y1 = *(const v4f*)(rowp + ko + 4);
      v4f y2 = *(const v4f*)(rowp + 16 + ko);
      v4f y3 = *(const v4f*)(rowp + 16 + ko + 4);

      // one set of loads feeds BOTH planes: P=[y==1], M=[y!=-1]
      v16h aP, aM;
#pragma unroll
      for (int e = 0; e < 4; ++e) {
        aP[e]      = (y0[e] == 1.0f)  ? h1 : h0;
        aM[e]      = (y0[e] != -1.0f) ? h1 : h0;
        aP[4 + e]  = (y1[e] == 1.0f)  ? h1 : h0;
        aM[4 + e]  = (y1[e] != -1.0f) ? h1 : h0;
        aP[8 + e]  = (y2[e] == 1.0f)  ? h1 : h0;
        aM[8 + e]  = (y2[e] != -1.0f) ? h1 : h0;
        aP[12 + e] = (y3[e] == 1.0f)  ? h1 : h0;
        aM[12 + e] = (y3[e] != -1.0f) ? h1 : h0;
      }

      // ---- B fragment, value chain: B[k, n=0]=alpha, B[k, n=1]=alpha^2 ----
      v16h bP;
      {
        const _Float16* src = (col == 1) ? alph2 : alph;
        v8h lo = *(const v8h*)(src + kb + ko);
        v8h hi = *(const v8h*)(src + kb + 16 + ko);
        _Float16 keep = (col < 2) ? h1 : h0;
#pragma unroll
        for (int i = 0; i < 8; ++i) { bP[i] = lo[i] * keep; bP[8 + i] = hi[i] * keep; }
      }

      accP = __builtin_amdgcn_wmma_f32_16x16x32_f16(false, aP, false, bP,
                                                    (short)0, accP, false, false);
      accM = __builtin_amdgcn_wmma_f32_16x16x32_f16(false, aM, false, bOnes,
                                                    (short)0, accM, false, false);
    }

    // D layout: lane holds N=col, VGPR v holds row M = v + 8*half16.
    // S1 lives in col==0 lanes, S2 in col==1 lanes, cnt in col==0 of accM.
    float tilePart = 0.0f;
#pragma unroll
    for (int v = 0; v < 8; ++v) {
      float s1  = accP[v];
      float s2  = __shfl(accP[v], (lane & 16) | 1, 32);  // pull col1 -> col0
      float cnt = accM[v];
      float di  = (cnt > 0.0f) ? (s2 / cnt - (s1 / cnt) * (s1 / cnt)) : 0.0f;
      if (col != 0) di = 0.0f;   // only col0 lanes carry valid row stats
      tilePart += di;
    }
    waveAcc += tilePart;
  }

  // butterfly wave reduction (non-col0 lanes contribute 0)
#pragma unroll
  for (int off = 16; off > 0; off >>= 1)
    waveAcc += __shfl_xor(waveAcc, off, 32);
  if (lane == 0) red[wave] = waveAcc;
  __syncthreads();
  if (tid == 0) {
    float s = 0.0f;
#pragma unroll
    for (int w = 0; w < WAVES_PER_BLOCK; ++w) s += red[w];
    blockPartials[blockIdx.x] = s;
  }
}

__global__ __launch_bounds__(BLOCK) void lm_final_kernel(
    const float* __restrict__ partials, int nPartials,
    const float* __restrict__ theta, float* __restrict__ out) {
  __shared__ float red[BLOCK];
  const int tid = threadIdx.x;
  float s = 0.0f;
  for (int i = tid; i < nPartials; i += BLOCK) s += partials[i];
  if (tid < KDIM) {
    float th = theta[tid];
    s += 1e-4f * th * th;
  }
  red[tid] = s;
  __syncthreads();
  for (int off = BLOCK / 2; off > 0; off >>= 1) {
    if (tid < off) red[tid] += red[tid + off];
    __syncthreads();
  }
  if (tid == 0) out[0] = red[0];
}

extern "C" void kernel_launch(void* const* d_in, const int* in_sizes, int n_in,
                              void* d_out, int out_size, void* d_ws, size_t ws_size,
                              hipStream_t stream) {
  const float* Y     = (const float*)d_in[0];   // [200000, 256] f32
  const float* theta = (const float*)d_in[1];   // [256] f32
  float* out      = (float*)d_out;              // scalar f32
  float* partials = (float*)d_ws;               // NBLOCKS floats of scratch

  lm_tile_kernel<<<NBLOCKS, BLOCK, 0, stream>>>(Y, theta, partials);
  lm_final_kernel<<<1, BLOCK, 0, stream>>>(partials, NBLOCKS, theta, out);
}
